// RNN_modelv2_47760036332020
// MI455X (gfx1250) — matrix operand order repaired and checked by reading, not compiled, not run on hardware
//
#include <hip/hip_runtime.h>

// ---------------------------------------------------------------------------
// Types for CDNA5 WMMA bf16 (v_wmma_f32_16x16x32_bf16) + TDM descriptors
// ---------------------------------------------------------------------------
typedef __attribute__((ext_vector_type(16))) __bf16 v16bf;
typedef __attribute__((ext_vector_type(8)))  __bf16 v8bf;
typedef __attribute__((ext_vector_type(8)))  float  v8f;
typedef __attribute__((ext_vector_type(4)))  unsigned int u32x4;
typedef __attribute__((ext_vector_type(8)))  int i32x8;
typedef __attribute__((ext_vector_type(4)))  int i32x4;

#define HID   2048
#define FOURH 8192
#define BATCH 16
#define TSEQ  128
#define MROWS (BATCH * TSEQ)   // 2048
#define SLAB  256              // K-slab staged into LDS by the TDM
#define NSLAB (HID / SLAB)     // 8

// ---------------------------------------------------------------------------
// WMMA fragment helpers (wave32, per CDNA5 ISA §7.12.2 layouts)
// ---------------------------------------------------------------------------
// A-matrix 16x32 bf16: lanes 0-15 row M=lane hold K=0..7 / K=16..23;
// lanes 16-31 hold K=8..15 / K=24..31 for the same rows.
__device__ __forceinline__ v16bf load_a_frag(const __bf16* __restrict__ base, int ld) {
  const int lane = threadIdx.x & 31;
  const int row  = lane & 15;
  const int kh   = (lane >> 4) * 8;          // 0 or 8
  const __bf16* p = base + row * ld + kh;
  v8bf lo = *reinterpret_cast<const v8bf*>(p);        // K = kh .. kh+7
  v8bf hi = *reinterpret_cast<const v8bf*>(p + 16);   // K = kh+16 .. kh+23
  v16bf r;
#pragma unroll
  for (int i = 0; i < 8; ++i) { r[i] = lo[i]; r[i + 8] = hi[i]; }
  return r;
}

// B-matrix 32x16 bf16 (B[k][n]): lane L<16 holds column N=L, K=0..7 / K=16..23;
// lanes 16-31 hold K=8..15 / K=24..31. W is row-major [n][k] so the fragment is
// two 8-k runs of one W row -> two contiguous 16B loads.
__device__ __forceinline__ v16bf load_b_frag(const __bf16* __restrict__ wbase, int ld) {
  const int lane = threadIdx.x & 31;
  const int col  = lane & 15;
  const int kh   = (lane >> 4) * 8;          // 0 or 8
  const __bf16* p = wbase + col * ld + kh;
  v8bf lo = *reinterpret_cast<const v8bf*>(p);
  v8bf hi = *reinterpret_cast<const v8bf*>(p + 16);
  v16bf r;
#pragma unroll
  for (int i = 0; i < 8; ++i) { r[i] = lo[i]; r[i + 8] = hi[i]; }
  return r;
}

__device__ __forceinline__ v8f wmma_bf16(v16bf a, v16bf b, v8f c) {
  return __builtin_amdgcn_wmma_f32_16x16x32_bf16(
      false, a, false, b, (short)0, c, false, false);
}

// ---------------------------------------------------------------------------
// TDM: DMA one 16-row x SLAB-col bf16 tile of h into LDS (D# per ISA §8.3/8.4).
// group0: count=1 | lds_addr | global_addr[31:0] | addr[56:32] + type=2
// group1: data_size=2B; tensor_dim0=HID, tensor_dim1=16; tile=SLABx16;
//         tensor_dim0_stride=HID. 2D tensor -> remaining groups zero.
// Toolchain here exposes the 6-arg builtin:
//   (u32x4 g0, i32x8 g1, i32x4 g2, i32x4 g3, i32x8 extra, i32 cpol)
// ---------------------------------------------------------------------------
__device__ __forceinline__ void tdm_load_h_slab(const __bf16* gsrc, unsigned lds_byte_off) {
  unsigned long long ga = (unsigned long long)(uintptr_t)gsrc;
  u32x4 g0;
  g0[0] = 1u;                                            // count=1 (valid D#)
  g0[1] = lds_byte_off;                                  // LDS dest (bytes)
  g0[2] = (unsigned)(ga & 0xFFFFFFFFu);
  g0[3] = (unsigned)((ga >> 32) & 0x01FFFFFFu) | 0x80000000u; // addr[56:32] | type=2
  i32x8 g1;
  g1[0] = (int)(1u << 16);                               // data_size = 2 bytes
  g1[1] = (int)((HID & 0xFFFFu) << 16);                  // tensor_dim0[15:0]
  g1[2] = (int)((HID >> 16) | (16u << 16));              // dim0[31:16] | tensor_dim1
  g1[3] = (int)((unsigned)SLAB << 16);                   // tile_dim0 = SLAB
  g1[4] = 16;                                            // tile_dim1 = 16 rows
  g1[5] = HID;                                           // tensor_dim0_stride
  g1[6] = 0;
  g1[7] = 0;
  i32x4 z4 = {0, 0, 0, 0};
  i32x8 z8 = {0, 0, 0, 0, 0, 0, 0, 0};
  __builtin_amdgcn_tensor_load_to_lds(g0, g1, z4, z4, z8, 0);
}

// ---------------------------------------------------------------------------
// Utility kernels
// ---------------------------------------------------------------------------
__global__ void cvt_f32_bf16_k(const float* __restrict__ in, __bf16* __restrict__ out, int n) {
  int i = blockIdx.x * blockDim.x + threadIdx.x;
  if (i < n) out[i] = (__bf16)in[i];
}

__global__ void bias_sum_k(const float* __restrict__ a, const float* __restrict__ b,
                           float* __restrict__ out, int n) {
  int i = blockIdx.x * blockDim.x + threadIdx.x;
  if (i < n) out[i] = a[i] + b[i];
}

__global__ void zero_bf16_k(__bf16* __restrict__ p, int n) {
  int i = blockIdx.x * blockDim.x + threadIdx.x;
  if (i < n) p[i] = (__bf16)0.f;
}

__global__ void zero_f32_k(float* __restrict__ p, int n) {
  int i = blockIdx.x * blockDim.x + threadIdx.x;
  if (i < n) p[i] = 0.f;
}

// ---------------------------------------------------------------------------
// Time-batched input projection: C[m,n] = sum_k A[m,k]*W[n,k] + bias[n]
// 2x2 register blocking: each wave owns a 32(M) x 32(N) patch -> per K-step
// 2 A frags + 2 B frags feed 4 WMMAs (load:math ratio 1:1).
// Grid: (N/256, M/32); block = 256 threads = 8 waves.
// ---------------------------------------------------------------------------
__global__ void __launch_bounds__(256)
xproj_gemm_k(const __bf16* __restrict__ A, const __bf16* __restrict__ W,
             const float* __restrict__ bias, float* __restrict__ C,
             int N, int K) {
  const int wave = threadIdx.x >> 5;
  const int n0 = blockIdx.x * 256 + wave * 32;
  const int m0 = blockIdx.y * 32;
  const __bf16* arow0 = A + (size_t)m0 * K;
  const __bf16* arow1 = A + (size_t)(m0 + 16) * K;
  const __bf16* wrow0 = W + (size_t)n0 * K;
  const __bf16* wrow1 = W + (size_t)(n0 + 16) * K;
  v8f acc00 = {}, acc01 = {}, acc10 = {}, acc11 = {};
#pragma unroll 2
  for (int k = 0; k < K; k += 32) {
    v16bf a0 = load_a_frag(arow0 + k, K);
    v16bf a1 = load_a_frag(arow1 + k, K);
    v16bf b0 = load_b_frag(wrow0 + k, K);
    v16bf b1 = load_b_frag(wrow1 + k, K);
    acc00 = wmma_bf16(a0, b0, acc00);
    acc01 = wmma_bf16(a0, b1, acc01);
    acc10 = wmma_bf16(a1, b0, acc10);
    acc11 = wmma_bf16(a1, b1, acc11);
  }
  const int lane = threadIdx.x & 31;
  const int nc = lane & 15;
  const int mb = (lane >> 4) * 8;
  const float bv0 = bias[n0 + nc];
  const float bv1 = bias[n0 + 16 + nc];
#pragma unroll
  for (int r = 0; r < 8; ++r) {
    C[(size_t)(m0 + mb + r) * N + (n0 + nc)]           = acc00[r] + bv0;
    C[(size_t)(m0 + mb + r) * N + (n0 + 16 + nc)]      = acc01[r] + bv1;
    C[(size_t)(m0 + 16 + mb + r) * N + (n0 + nc)]      = acc10[r] + bv0;
    C[(size_t)(m0 + 16 + mb + r) * N + (n0 + 16 + nc)] = acc11[r] + bv1;
  }
}

// ---------------------------------------------------------------------------
// One LSTM timestep. Block owns 32 hidden columns j0..j0+31; 8 waves compute
// the i/f/g/o 16x16 gate tiles. h_{t-1} (64KB, shared by all waves) is staged
// into LDS in double-buffered SLAB-wide tiles by the Tensor Data Mover
// (tensor_load_to_lds + s_wait_tensorcnt), W_hh streams from L2 (bf16 weights
// fit the 192MB L2 across all 128 steps). Gate tiles exchanged via LDS, then
// the fused sigmoid/tanh cell update. h is double-buffered across timesteps.
// ---------------------------------------------------------------------------
__global__ void __launch_bounds__(256)
lstm_step_k(const __bf16* __restrict__ Whh,    // [4H, H] bf16
            const float*  __restrict__ xproj,  // [B, T, 4H] fp32 (bias folded in)
            const __bf16* __restrict__ h_in,   // [B, H] bf16
            __bf16*       __restrict__ h_out,  // [B, H] bf16
            float*        __restrict__ c_state,// [B, H] fp32
            __bf16*       __restrict__ seq_out,// [B, T, H] bf16
            int t) {
  // hsm declared first -> LDS offset 0 (TDM descriptor assumes this).
  __shared__ __bf16 hsm[2][BATCH][SLAB];       // 16 KB: double-buffered h slabs
  __shared__ float  gsm[8][16][16];            // 8 KB: gate tile exchange

  const int wave = threadIdx.x >> 5;
  const int lane = threadIdx.x & 31;
  const int j0   = blockIdx.x * 32;
  const int gate = wave >> 1;
  const int half = wave & 1;
  const int n0   = gate * HID + j0 + half * 16;

  const __bf16* wrow = Whh + (size_t)n0 * HID;

  if (wave == 0) tdm_load_h_slab(h_in, 0u);    // prime slab 0

  v8f acc = {};
  for (int s = 0; s < NSLAB; ++s) {
    __syncthreads();                           // prev compute done; buffers free
    if (wave == 0) {
      if (s + 1 < NSLAB) {
        tdm_load_h_slab(h_in + (s + 1) * SLAB, (unsigned)(((s + 1) & 1) * sizeof(hsm[0])));
        __builtin_amdgcn_s_wait_tensorcnt(1);  // slab s landed
      } else {
        __builtin_amdgcn_s_wait_tensorcnt(0);  // last slab landed
      }
    }
    __syncthreads();                           // slab s visible to all waves

    const __bf16* hbase = &hsm[s & 1][0][0];
    const __bf16* wbase = wrow + s * SLAB;
#pragma unroll 4
    for (int kk = 0; kk < SLAB; kk += 32) {
      if (kk + 128 < SLAB)
        __builtin_prefetch(wbase + ((lane & 15) * HID) + kk + 128, 0, 0);
      v16bf a = load_a_frag(hbase + kk, SLAB); // ds_load from staged h
      v16bf b = load_b_frag(wbase + kk, HID);  // L2-resident W_hh stream
      acc = wmma_bf16(a, b, acc);
    }
  }

  // add x_proj contribution and park gate tile in LDS (C layout: lane holds
  // n = lane&15, m = (lane>>4)*8 + r)
  const int nn = n0 + (lane & 15);
  const int mb = (lane >> 4) * 8;
#pragma unroll
  for (int r = 0; r < 8; ++r) {
    const int b = mb + r;
    gsm[wave][b][lane & 15] = acc[r] + xproj[((size_t)b * TSEQ + t) * FOURH + nn];
  }
  __syncthreads();

  // fused elementwise cell update: 16 batches x 32 cols = 512 elems
  for (int e = threadIdx.x; e < 512; e += 256) {
    const int b  = e >> 5;        // 0..15
    const int jj = e & 31;        // 0..31
    const int hh = jj >> 4;       // which half-tile
    const int cc = jj & 15;
    float iv = gsm[0 + hh][b][cc];
    float fv = gsm[2 + hh][b][cc];
    float gv = gsm[4 + hh][b][cc];
    float ov = gsm[6 + hh][b][cc];
    iv = 1.f / (1.f + __expf(-iv));
    fv = 1.f / (1.f + __expf(-fv));
    gv = tanhf(gv);
    ov = 1.f / (1.f + __expf(-ov));
    const int j = j0 + jj;
    float c = fv * c_state[b * HID + j] + iv * gv;
    float h = ov * tanhf(c);
    c_state[b * HID + j] = c;
    const __bf16 hb = (__bf16)h;
    h_out[b * HID + j] = hb;
    seq_out[((size_t)b * TSEQ + t) * HID + j] = hb;
  }
}

// ---------------------------------------------------------------------------
// Final projection: out[m] = dot(seq[m,:], W_out[0,:]) + b_out ; one block per m
// ---------------------------------------------------------------------------
__global__ void __launch_bounds__(256)
out_proj_k(const __bf16* __restrict__ seq, const float* __restrict__ Wout,
           const float* __restrict__ bout, float* __restrict__ out) {
  const int m = blockIdx.x;
  const __bf16* row = seq + (size_t)m * HID;
  float s = 0.f;
  for (int j = threadIdx.x; j < HID; j += 256) s += (float)row[j] * Wout[j];
#pragma unroll
  for (int off = 16; off > 0; off >>= 1) s += __shfl_down(s, off, 32);
  __shared__ float red[8];
  if ((threadIdx.x & 31) == 0) red[threadIdx.x >> 5] = s;
  __syncthreads();
  if (threadIdx.x == 0) {
    float tot = 0.f;
#pragma unroll
    for (int w = 0; w < 8; ++w) tot += red[w];
    out[m] = tot + bout[0];
  }
}

// ---------------------------------------------------------------------------
// Host orchestration
// ---------------------------------------------------------------------------
static inline int cdiv(int a, int b) { return (a + b - 1) / b; }

extern "C" void kernel_launch(void* const* d_in, const int* in_sizes, int n_in,
                              void* d_out, int out_size, void* d_ws, size_t ws_size,
                              hipStream_t stream) {
  (void)in_sizes; (void)n_in; (void)out_size; (void)ws_size;
  const float* x    = (const float*)d_in[0];
  const float* Wout = (const float*)d_in[13];
  const float* bout = (const float*)d_in[14];
  float* out = (float*)d_out;

  // -------- workspace carve-up (~151 MB) --------
  char* ws = (char*)d_ws;
  size_t off = 0;
  auto carve = [&](size_t bytes) -> void* {
    void* p = ws + off;
    off = (off + bytes + 255) & ~(size_t)255;
    return p;
  };
  __bf16* wih_bf = (__bf16*)carve((size_t)FOURH * HID * 2);   // 33.5 MB (max K)
  __bf16* whh_bf = (__bf16*)carve((size_t)FOURH * HID * 2);   // 33.5 MB
  float*  bias   = (float*) carve((size_t)FOURH * 4);
  float*  xproj  = (float*) carve((size_t)MROWS * FOURH * 4); // 67 MB
  __bf16* seqA   = (__bf16*)carve((size_t)MROWS * HID * 2);   // 8.4 MB
  __bf16* seqB   = (__bf16*)carve((size_t)MROWS * HID * 2);   // 8.4 MB
  __bf16* h0     = (__bf16*)carve((size_t)BATCH * HID * 2);
  __bf16* h1     = (__bf16*)carve((size_t)BATCH * HID * 2);
  float*  cst    = (float*) carve((size_t)BATCH * HID * 4);

  const int Kin[3] = {128, HID, HID};

  // layer-0 input: fp32 [B,T,128] -> bf16 rows of length 128
  {
    int n = MROWS * 128;
    cvt_f32_bf16_k<<<cdiv(n, 256), 256, 0, stream>>>(x, seqA, n);
  }

  __bf16* seq_in  = seqA;
  __bf16* seq_out = seqB;

  for (int l = 0; l < 3; ++l) {
    const float* Wih = (const float*)d_in[1 + 4 * l];
    const float* Whh = (const float*)d_in[2 + 4 * l];
    const float* bih = (const float*)d_in[3 + 4 * l];
    const float* bhh = (const float*)d_in[4 + 4 * l];
    const int K = Kin[l];

    { int n = FOURH * K;   cvt_f32_bf16_k<<<cdiv(n, 256), 256, 0, stream>>>(Wih, wih_bf, n); }
    { int n = FOURH * HID; cvt_f32_bf16_k<<<cdiv(n, 256), 256, 0, stream>>>(Whh, whh_bf, n); }
    bias_sum_k<<<cdiv(FOURH, 256), 256, 0, stream>>>(bih, bhh, bias, FOURH);

    // x_proj for all timesteps: [2048, K] x [8192, K]^T -> [2048, 8192]
    dim3 g(FOURH / 256, MROWS / 32);
    xproj_gemm_k<<<g, 256, 0, stream>>>(seq_in, wih_bf, bias, xproj, FOURH, K);

    { int n = BATCH * HID;
      zero_bf16_k<<<cdiv(n, 256), 256, 0, stream>>>(h0, n);
      zero_f32_k <<<cdiv(n, 256), 256, 0, stream>>>(cst, n); }

    __bf16* hin = h0;
    __bf16* hout = h1;
    for (int t = 0; t < TSEQ; ++t) {
      lstm_step_k<<<HID / 32, 256, 0, stream>>>(whh_bf, xproj, hin, hout, cst, seq_out, t);
      __bf16* tmp = hin; hin = hout; hout = tmp;
    }

    __bf16* s = seq_in; seq_in = seq_out; seq_out = s;
  }

  // final projection to [B, T, 1]
  out_proj_k<<<MROWS, 256, 0, stream>>>(seq_in, Wout, bout, out);
}
